// DecoderWithAttention_8392366097012
// MI455X (gfx1250) — matrix operand order repaired and checked by reading, not compiled, not run on hardware
//
#include <hip/hip_runtime.h>
#include <math.h>

// ---------------- problem constants ----------------
#define NB   64      // batch
#define NP   196     // pixels
#define NE   2048    // encoder dim
#define NL   32      // max caption length
#define NV   10000   // vocab
#define NDEC 512     // decoder dim (== EMB == ATT)
#define NT   31      // decode steps (L-1)

typedef __attribute__((ext_vector_type(16))) __bf16        v16bf;
typedef __attribute__((ext_vector_type(8)))  float         v8f;
typedef __attribute__((ext_vector_type(8)))  unsigned int  v8ui;

__device__ __forceinline__ unsigned int pack_bf16(float lo, float hi)
{
    unsigned short a = __builtin_bit_cast(unsigned short, (__bf16)lo);
    unsigned short b = __builtin_bit_cast(unsigned short, (__bf16)hi);
    return ((unsigned int)b << 16) | (unsigned int)a;
}

// Async DMA 16B global -> LDS (ASYNCcnt path, §15.18.3 GLOBAL_LOAD_ASYNC_TO_LDS_B128).
// Generic __shared__ pointers carry the LDS byte offset in their low 32 bits
// (LDS aperture truncates addr[31:0]).
__device__ __forceinline__ void async_copy16(const unsigned int* gsrc, unsigned int* lds)
{
    unsigned int loff = (unsigned int)(size_t)lds;
    unsigned long long gaddr = (unsigned long long)(size_t)gsrc;
    asm volatile("global_load_async_to_lds_b128 %0, %1, off"
                 :: "v"(loff), "v"(gaddr) : "memory");
}
__device__ __forceinline__ void async_wait0()
{
    asm volatile("s_wait_asynccnt 0" ::: "memory");
}

// ================= generic bf16-WMMA GEMM =================
// C[m,n] = (ACCUM? C : 0) + A[m,:K] @ B[:K,n] + bias[n]
// A: f32 row-major (converted to bf16 in LDS).
// Bp: PRE-PACKED bf16 pair rows — uint32 Bp[k/2 * N + n] = (B[k][n], B[k+1][n]).
// REMAP: row written at row_remap[m], result scaled by row_scale[m].
// Requirements: M % 64 == 0, N % 16 == 0, K % 64 == 0, rows 16B-aligned.
// Block: 128 threads (4 waves). Block tile: 64(M) x 32(N), K-step 64.
#define GBM 64
#define GBK 64
#define SAU 36   // A tile LDS row stride in uints
#define SBU 36   // B tile LDS pair-row stride in uints (multiple of 4 -> 16B-aligned rows)

template<bool ACCUM, bool REMAP>
__global__ __launch_bounds__(128)
void gemm_wmma_bf16(const float* __restrict__ A, int lda,
                    const unsigned int* __restrict__ Bp,
                    const float* __restrict__ bias,
                    float* __restrict__ C, int ldc,
                    int K, int N,
                    const int*   __restrict__ row_remap,
                    const float* __restrict__ row_scale)
{
    __shared__ __align__(16) unsigned int Asp[GBM * SAU];  // A: 64 rows x 32 k-pairs
    __shared__ __align__(16) unsigned int Bsp[32 * SBU];   // B: 32 k-pairs x 32 cols

    const int tid  = threadIdx.x;
    const int wave = tid >> 5;
    const int lane = tid & 31;
    const int m0   = blockIdx.y * GBM;
    const int n0   = blockIdx.x * 32;
    const bool has2 = (n0 + 16) < N;          // block-uniform

    v8f acc0 = {}, acc1 = {};
    const int mrow  = lane & 15;   // A-frag: M = lane&15 (both half-waves)
    const int khalf = lane >> 4;   // A-frag: K half selector
    const int bn    = lane & 15;   // B-frag: column
    const int ksel  = lane >> 4;   // B-frag: K half selector

    for (int k0 = 0; k0 < K; k0 += GBK) {
        // --- B tile: async DMA 32 kpair-rows x 32 packed cols, global -> LDS ---
        {
            const unsigned int* bsrc = Bp + (size_t)(k0 >> 1) * N + n0;
#pragma unroll
            for (int i = 0; i < 2; ++i) {
                int q  = tid + 128 * i;      // 256 x 16B chunks
                int kp = q >> 3;
                int c4 = (q & 7) << 2;
                if (n0 + c4 < N)
                    async_copy16(bsrc + (size_t)kp * N + c4, &Bsp[kp * SBU + c4]);
            }
        }
        // --- A tile: 64x64 f32 -> packed bf16 pairs in LDS (conversion path) ---
#pragma unroll
        for (int i = 0; i < 8; ++i) {
            int q  = tid + 128 * i;          // 1024 float4 loads
            int r  = q >> 4;
            int c4 = (q & 15) << 2;
            const float4 v = *(const float4*)(A + (size_t)(m0 + r) * lda + k0 + c4);
            unsigned int* dst = &Asp[r * SAU + (c4 >> 1)];
            dst[0] = pack_bf16(v.x, v.y);
            dst[1] = pack_bf16(v.z, v.w);
        }
        async_wait0();
        __syncthreads();

        const unsigned int* arow = &Asp[(wave * 16 + mrow) * SAU];
#pragma unroll
        for (int kk = 0; kk < 2; ++kk) {     // two 32-wide K chunks
            // A fragment: contiguous uints -> 2x ds_load_b128
            v8ui au;
#pragma unroll
            for (int r = 0; r < 8; ++r) {
                int ku = (kk << 4) + (khalf << 3) + r;
                au[r] = arow[ku];
            }
            v16bf af = __builtin_bit_cast(v16bf, au);
            // B fragments (packed pair rows)
            v8ui bu0;
#pragma unroll
            for (int r = 0; r < 8; ++r)
                bu0[r] = Bsp[((kk << 4) + (ksel << 3) + r) * SBU + bn];
            v16bf bf0 = __builtin_bit_cast(v16bf, bu0);
            acc0 = __builtin_amdgcn_wmma_f32_16x16x32_bf16(
                       false, af, false, bf0, (short)0, acc0, false, false);
            if (has2) {
                v8ui bu1;
#pragma unroll
                for (int r = 0; r < 8; ++r)
                    bu1[r] = Bsp[((kk << 4) + (ksel << 3) + r) * SBU + 16 + bn];
                v16bf bf1 = __builtin_bit_cast(v16bf, bu1);
                acc1 = __builtin_amdgcn_wmma_f32_16x16x32_bf16(
                           false, af, false, bf1, (short)0, acc1, false, false);
            }
        }
        __syncthreads();
    }

    // --- branch-free epilogue (template-specialized) ---
    const int nA = n0 + bn;
    const float bv0 = bias[nA];
    const float bv1 = has2 ? bias[nA + 16] : 0.0f;
#pragma unroll
    for (int r = 0; r < 8; ++r) {
        int m = m0 + wave * 16 + r + ((lane >> 4) << 3);
        int orow  = REMAP ? row_remap[m] : m;
        float scl = REMAP ? row_scale[m] : 1.0f;
        float* cp0 = C + (size_t)orow * ldc + nA;
        float v0 = acc0[r] + bv0;
        if (ACCUM) v0 += cp0[0];
        if (REMAP) v0 *= scl;
        cp0[0] = v0;
        if (has2) {
            float* cp1 = cp0 + 16;
            float v1 = acc1[r] + bv1;
            if (ACCUM) v1 += cp1[0];
            if (REMAP) v1 *= scl;
            cp1[0] = v1;
        }
    }
}

// ================= weight prepack: f32 (K x N) -> uint32 (K/2 x N) bf16 pairs =====
__global__ void prepack_kernel(const float* __restrict__ W, unsigned int* __restrict__ Wp,
                               int Kp, int N)
{
    int idx = blockIdx.x * blockDim.x + threadIdx.x;
    if (idx >= Kp * N) return;
    int kp = idx / N, n = idx - kp * N;
    const float* p = W + (size_t)(2 * kp) * N + n;
    Wp[idx] = pack_bf16(p[0], p[N]);
}

// ================= helper kernels =================
__device__ __forceinline__ float sigf(float x) { return 1.0f / (1.0f + __expf(-x)); }

// Stable descending rank of caption lengths (inverse of argsort(-lens)).
__global__ void prep_kernel(const int* __restrict__ lens,
                            int* __restrict__ rank, int* __restrict__ dlen)
{
    int i = threadIdx.x;
    if (i >= NB) return;
    int li = lens[i];
    int r = 0;
    for (int j = 0; j < NB; ++j) {
        int lj = lens[j];
        if (lj > li || (lj == li && j < i)) ++r;
    }
    rank[i] = r;
    dlen[i] = li - 1;
}

// mean over P of encoder_out (original batch order)
__global__ void mean_kernel(const float* __restrict__ enc, float* __restrict__ out)
{
    int idx = blockIdx.x * blockDim.x + threadIdx.x;     // NB*NE
    if (idx >= NB * NE) return;
    int b = idx >> 11, e = idx & (NE - 1);
    const float* p = enc + (size_t)b * NP * NE + e;
    float s = 0.0f;
    for (int j = 0; j < NP; ++j) s += p[(size_t)j * NE];
    out[idx] = s * (1.0f / (float)NP);
}

// e[b,p] = relu(att1[b,p,:] + att2[b,:]) . Wf + bf   (one wave per (b,p))
__global__ __launch_bounds__(256)
void attn_e_kernel(const float* __restrict__ att1, const float* __restrict__ att2,
                   const float* __restrict__ Wf, const float* __restrict__ bfp,
                   float* __restrict__ e)
{
    int row  = blockIdx.x * 8 + (threadIdx.x >> 5);
    int lane = threadIdx.x & 31;
    if (row >= NB * NP) return;
    int b = row / NP;
    const float* a1 = att1 + (size_t)row * NDEC;
    const float* a2 = att2 + (size_t)b * NDEC;
    float s = 0.0f;
    for (int j = lane; j < NDEC; j += 32) {
        float v = a1[j] + a2[j];
        v = v > 0.0f ? v : 0.0f;
        s += v * Wf[j];
    }
#pragma unroll
    for (int off = 16; off > 0; off >>= 1) s += __shfl_xor(s, off, 32);
    if (lane == 0) e[row] = s + bfp[0];
}

// softmax over P + context vector; writes masked alphas to output.
__global__ __launch_bounds__(256)
void softmax_awe_kernel(const float* __restrict__ e, const float* __restrict__ enc,
                        const int* __restrict__ rank, const int* __restrict__ dlen,
                        float* __restrict__ awe, float* __restrict__ alphas_out, int t)
{
    __shared__ float sh[256];
    __shared__ float sa[NP];
    int b = blockIdx.x, tid = threadIdx.x;
    float v = (tid < NP) ? e[b * NP + tid] : -3.0e38f;
    sh[tid] = v; __syncthreads();
    for (int s = 128; s > 0; s >>= 1) {
        if (tid < s) sh[tid] = fmaxf(sh[tid], sh[tid + s]);
        __syncthreads();
    }
    float mx = sh[0]; __syncthreads();
    float ex = (tid < NP) ? __expf(v - mx) : 0.0f;
    sh[tid] = ex; __syncthreads();
    for (int s = 128; s > 0; s >>= 1) {
        if (tid < s) sh[tid] += sh[tid + s];
        __syncthreads();
    }
    float alpha = ex / sh[0];
    if (tid < NP) sa[tid] = alpha;
    float mask = (t < dlen[b]) ? 1.0f : 0.0f;
    if (tid < NP)
        alphas_out[((size_t)rank[b] * NT + t) * NP + tid] = alpha * mask;
    __syncthreads();
    const float* eb = enc + (size_t)b * NP * NE;
    for (int k = 0; k < NE / 256; ++k) {
        int ei = tid + 256 * k;
        float s = 0.0f;
        for (int p = 0; p < NP; ++p) s += sa[p] * eb[(size_t)p * NE + ei];
        awe[(size_t)b * NE + ei] = s;
    }
}

// awe *= sigmoid(gate_lin)
__global__ void gate_kernel(float* __restrict__ awe, const float* __restrict__ glin)
{
    int idx = blockIdx.x * blockDim.x + threadIdx.x;
    if (idx >= NB * NE) return;
    awe[idx] *= sigf(glin[idx]);
}

// x = concat(emb[caps[b,t]], awe[b])
__global__ void xbuild_kernel(const int* __restrict__ caps, const float* __restrict__ emb,
                              const float* __restrict__ awe, float* __restrict__ x, int t)
{
    int idx = blockIdx.x * blockDim.x + threadIdx.x;     // NB*(NDEC+NE)
    if (idx >= NB * (NDEC + NE)) return;
    int b = idx / (NDEC + NE), j = idx % (NDEC + NE);
    float v;
    if (j < NDEC) {
        int tok = caps[b * NL + t];
        v = emb[(size_t)tok * NDEC + j];
    } else {
        v = awe[(size_t)b * NE + (j - NDEC)];
    }
    x[idx] = v;
}

// LSTM cell pointwise + masked state update; also emits per-row mask.
__global__ void lstm_kernel(const float* __restrict__ g, float* __restrict__ h,
                            float* __restrict__ c, float* __restrict__ hn,
                            const int* __restrict__ dlen, float* __restrict__ maskf, int t)
{
    int idx = blockIdx.x * blockDim.x + threadIdx.x;     // NB*NDEC
    if (idx >= NB * NDEC) return;
    int b = idx >> 9, d = idx & (NDEC - 1);
    const float* gb = g + (size_t)b * 4 * NDEC;
    float ig = sigf(gb[d]);
    float fg = sigf(gb[NDEC + d]);
    float gg = tanhf(gb[2 * NDEC + d]);
    float og = sigf(gb[3 * NDEC + d]);
    float cn = fg * c[idx] + ig * gg;
    float hv = og * tanhf(cn);
    hn[idx] = hv;
    float mask = (t < dlen[b]) ? 1.0f : 0.0f;
    h[idx] = mask * hv + (1.0f - mask) * h[idx];
    c[idx] = mask * cn + (1.0f - mask) * c[idx];
    if (d == 0) maskf[b] = mask;
}

// ================= host orchestration =================
static inline size_t align256(size_t x) { return (x + 255) & ~(size_t)255; }

extern "C" void kernel_launch(void* const* d_in, const int* in_sizes, int n_in,
                              void* d_out, int out_size, void* d_ws, size_t ws_size,
                              hipStream_t stream)
{
    (void)in_sizes; (void)n_in; (void)out_size; (void)ws_size;

    const float* enc   = (const float*)d_in[0];   // (B,P,E)
    const int*   caps  = (const int*)  d_in[1];   // (B,L)
    const int*   lens  = (const int*)  d_in[2];   // (B,1)
    const float* emb   = (const float*)d_in[3];   // (V,EMB)
    const float* We    = (const float*)d_in[4];   // (E,ATT)
    const float* be    = (const float*)d_in[5];
    const float* Wd    = (const float*)d_in[6];   // (DEC,ATT)
    const float* bd    = (const float*)d_in[7];
    const float* Wf    = (const float*)d_in[8];   // (ATT,1)
    const float* bfp   = (const float*)d_in[9];
    const float* Wih   = (const float*)d_in[10];  // (2560,2048)
    const float* bih   = (const float*)d_in[11];
    const float* Whh   = (const float*)d_in[12];  // (512,2048)
    const float* bhh   = (const float*)d_in[13];
    const float* Wb    = (const float*)d_in[14];  // (512,2048)
    const float* bb    = (const float*)d_in[15];
    const float* Whi   = (const float*)d_in[16];  // (2048,512)
    const float* bhi   = (const float*)d_in[17];
    const float* Wci   = (const float*)d_in[18];
    const float* bci   = (const float*)d_in[19];
    const float* Wfc   = (const float*)d_in[20];  // (512,10000)
    const float* bfc   = (const float*)d_in[21];

    float* out_pred  = (float*)d_out;                                  // (B,T,V)
    float* out_alpha = (float*)d_out + (size_t)NB * NT * NV;           // (B,T,P)

    // ---- workspace carve-up ----
    char* w = (char*)d_ws;
    size_t off = 0;
    auto carve = [&](size_t bytes) { char* p = w + off; off = align256(off + bytes); return p; };
    int*   rank     = (int*)  carve(NB * sizeof(int));
    int*   dlen     = (int*)  carve(NB * sizeof(int));
    float* maskf    = (float*)carve(NB * sizeof(float));
    float* mean_enc = (float*)carve((size_t)NB * NE * sizeof(float));
    float* h        = (float*)carve((size_t)NB * NDEC * sizeof(float));
    float* c        = (float*)carve((size_t)NB * NDEC * sizeof(float));
    float* hn       = (float*)carve((size_t)NB * NDEC * sizeof(float));
    float* att2     = (float*)carve((size_t)NB * NDEC * sizeof(float));
    float* ebuf     = (float*)carve((size_t)NB * NP * sizeof(float));
    float* awe      = (float*)carve((size_t)NB * NE * sizeof(float));
    float* glin     = (float*)carve((size_t)NB * NE * sizeof(float));
    float* gbuf     = (float*)carve((size_t)NB * 4 * NDEC * sizeof(float));
    float* xbuf     = (float*)carve((size_t)NB * (NDEC + NE) * sizeof(float));
    float* att1     = (float*)carve((size_t)NB * NP * NDEC * sizeof(float));  // 25.7 MB
    // packed bf16-pair weights (uint32 per K-pair)
    auto carveU = [&](size_t n) { return (unsigned int*)carve(n * sizeof(unsigned int)); };
    unsigned int* We_p  = carveU((size_t)(NE / 2) * NDEC);            // 1024x512
    unsigned int* Wd_p  = carveU((size_t)(NDEC / 2) * NDEC);          // 256x512
    unsigned int* Wb_p  = carveU((size_t)(NDEC / 2) * NE);            // 256x2048
    unsigned int* Wih_p = carveU((size_t)((NDEC + NE) / 2) * 4 * NDEC); // 1280x2048
    unsigned int* Whh_p = carveU((size_t)(NDEC / 2) * 4 * NDEC);      // 256x2048
    unsigned int* Whi_p = carveU((size_t)(NE / 2) * NDEC);            // 1024x512
    unsigned int* Wci_p = carveU((size_t)(NE / 2) * NDEC);            // 1024x512
    unsigned int* Wfc_p = carveU((size_t)(NDEC / 2) * NV);            // 256x10000

    auto gx = [](int N) { return (N + 31) / 32; };
    auto prepack = [&](const float* W, unsigned int* Wp, int K, int N) {
        int total = (K / 2) * N;
        prepack_kernel<<<(total + 255) / 256, 256, 0, stream>>>(W, Wp, K / 2, N);
    };

    // ---- prologue ----
    prep_kernel<<<1, 64, 0, stream>>>(lens, rank, dlen);
    mean_kernel<<<(NB * NE) / 256, 256, 0, stream>>>(enc, mean_enc);
    prepack(We,  We_p,  NE,        NDEC);
    prepack(Wd,  Wd_p,  NDEC,      NDEC);
    prepack(Wb,  Wb_p,  NDEC,      NE);
    prepack(Wih, Wih_p, NDEC + NE, 4 * NDEC);
    prepack(Whh, Whh_p, NDEC,      4 * NDEC);
    prepack(Whi, Whi_p, NE,        NDEC);
    prepack(Wci, Wci_p, NE,        NDEC);
    prepack(Wfc, Wfc_p, NDEC,      NV);

    // h0 = mean_enc @ Whi + bhi ; c0 = mean_enc @ Wci + bci     (64x2048x512)
    gemm_wmma_bf16<false, false><<<dim3(gx(NDEC), 1), 128, 0, stream>>>(
        mean_enc, NE, Whi_p, bhi, h, NDEC, NE, NDEC, nullptr, nullptr);
    gemm_wmma_bf16<false, false><<<dim3(gx(NDEC), 1), 128, 0, stream>>>(
        mean_enc, NE, Wci_p, bci, c, NDEC, NE, NDEC, nullptr, nullptr);

    // att1 = enc @ We + be     (12544x2048x512) — enc in original order
    gemm_wmma_bf16<false, false><<<dim3(gx(NDEC), (NB * NP) / GBM), 128, 0, stream>>>(
        enc, NE, We_p, be, att1, NDEC, NE, NDEC, nullptr, nullptr);

    // ---- sequential decode ----
    for (int t = 0; t < NT; ++t) {
        // att2 = h @ Wd + bd                               (64x512x512)
        gemm_wmma_bf16<false, false><<<dim3(gx(NDEC), 1), 128, 0, stream>>>(
            h, NDEC, Wd_p, bd, att2, NDEC, NDEC, NDEC, nullptr, nullptr);

        // e = relu(att1 + att2) . Wf + bf
        attn_e_kernel<<<(NB * NP) / 8, 256, 0, stream>>>(att1, att2, Wf, bfp, ebuf);

        // alpha = softmax(e); awe = sum_p alpha*enc; write masked alphas
        softmax_awe_kernel<<<NB, 256, 0, stream>>>(ebuf, enc, rank, dlen, awe, out_alpha, t);

        // gate = sigmoid(h @ Wb + bb); awe *= gate         (64x512x2048)
        gemm_wmma_bf16<false, false><<<dim3(gx(NE), 1), 128, 0, stream>>>(
            h, NDEC, Wb_p, bb, glin, NE, NDEC, NE, nullptr, nullptr);
        gate_kernel<<<(NB * NE) / 256, 256, 0, stream>>>(awe, glin);

        // x = [emb[caps[:,t]], awe]
        xbuild_kernel<<<(NB * (NDEC + NE)) / 256, 256, 0, stream>>>(caps, emb, awe, xbuf, t);

        // g = x @ Wih + bih                                (64x2560x2048)
        gemm_wmma_bf16<false, false><<<dim3(gx(4 * NDEC), 1), 128, 0, stream>>>(
            xbuf, NDEC + NE, Wih_p, bih, gbuf, 4 * NDEC, NDEC + NE, 4 * NDEC,
            nullptr, nullptr);
        // g += h @ Whh + bhh                               (64x512x2048, accumulate)
        gemm_wmma_bf16<true, false><<<dim3(gx(4 * NDEC), 1), 128, 0, stream>>>(
            h, NDEC, Whh_p, bhh, gbuf, 4 * NDEC, NDEC, 4 * NDEC,
            nullptr, nullptr);

        // LSTM pointwise, masked state update
        lstm_kernel<<<(NB * NDEC) / 256, 256, 0, stream>>>(gbuf, h, c, hn, dlen, maskf, t);

        // preds[rank[b], t, :] = (hn @ Wfc + bfc) * mask   (64x512x10000)
        gemm_wmma_bf16<false, true><<<dim3(gx(NV), 1), 128, 0, stream>>>(
            hn, NDEC, Wfc_p, bfc, out_pred + (size_t)t * NV, NT * NV, NDEC, NV,
            rank, maskf);
    }
}
